// SwinTransformerBlock_609885356507
// MI455X (gfx1250) — matrix operand order repaired and checked by reading, not compiled, not run on hardware
//
#include <hip/hip_runtime.h>
#include <hip/hip_bf16.h>
#include <math.h>

typedef _Float16 f16;
typedef __attribute__((ext_vector_type(16))) _Float16 v16h;
typedef __attribute__((ext_vector_type(8)))  _Float16 v8h;
typedef __attribute__((ext_vector_type(8)))  float    v8f;

#define WMMA_F16(a,b,c) __builtin_amdgcn_wmma_f32_16x16x32_f16(false,(a),false,(b),(short)0,(c),false,false)

// Problem constants
constexpr int    C      = 384;
constexpr int    NHEAD  = 12;
constexpr int    HD     = 32;      // head dim
constexpr int    HID    = 1536;
constexpr int    SHIFT  = 4;
constexpr int    MROWS  = 32 * 64 * 64;          // 131072 tokens
constexpr size_t S0E    = (size_t)MROWS * C;     // elems of one [M,C] f16 tensor
constexpr float  QSCALE = 0.17677669529663689f;  // 1/sqrt(32)

__device__ __forceinline__ int regid(int h) { return h < 56 ? 0 : (h < 60 ? 1 : 2); }

// Map window-token row -> global token index (shift + window partition, also the
// inverse scatter for window-reverse + un-shift).
__device__ __forceinline__ int swin_map(int rr) {
    int win = rr >> 6, t = rr & 63;
    int b  = win >> 6, wl = win & 63;
    int wh = wl >> 3,  ww = wl & 7;
    int i  = t >> 3,   j  = t & 7;
    int h  = ((wh << 3) + i + SHIFT) & 63;
    int w  = ((ww << 3) + j + SHIFT) & 63;
    return (b << 12) + (h << 6) + w;
}

__device__ __forceinline__ v16h mk16(v8h lo, v8h hi) {
    return __builtin_shufflevector(lo, hi, 0,1,2,3,4,5,6,7,8,9,10,11,12,13,14,15);
}
__device__ __forceinline__ v16h ld16(const f16* p0, const f16* p1) {
    return mk16(*(const v8h*)p0, *(const v8h*)p1);
}

// ---------------------------------------------------------------------------
// Weight transpose + f32 -> f16: src[K,N] -> dst[N,K]
// ---------------------------------------------------------------------------
__global__ void transpose_f32_f16(const float* __restrict__ src, f16* __restrict__ dst,
                                  int Kd, int Nd) {
    int idx = blockIdx.x * 256 + threadIdx.x;
    if (idx >= Kd * Nd) return;
    int k = idx / Nd, n = idx % Nd;
    dst[(size_t)n * Kd + k] = (f16)src[(size_t)k * Nd + n];
}

// ---------------------------------------------------------------------------
// LayerNorm (128 threads, 3 channels/thread), optional swin gather mapping.
// ---------------------------------------------------------------------------
__device__ __forceinline__ float blockReduce128(float v, float* s) {
    #pragma unroll
    for (int o = 16; o > 0; o >>= 1) v += __shfl_down(v, o);
    __syncthreads();
    if ((threadIdx.x & 31) == 0) s[threadIdx.x >> 5] = v;
    __syncthreads();
    return s[0] + s[1] + s[2] + s[3];
}

__global__ __launch_bounds__(128)
void ln_kernel(const float* __restrict__ src, const float* __restrict__ g,
               const float* __restrict__ b, f16* __restrict__ dst, int mode) {
    __shared__ float sred[4];
    int r = blockIdx.x;
    int gs = mode ? swin_map(r) : r;
    const float* sp = src + (size_t)gs * C;
    int t = threadIdx.x;
    float x0 = sp[t], x1 = sp[t + 128], x2 = sp[t + 256];
    float total = blockReduce128(x0 + x1 + x2, sred);
    float mean = total * (1.0f / 384.0f);
    float d0 = x0 - mean, d1 = x1 - mean, d2 = x2 - mean;
    float vs = blockReduce128(d0 * d0 + d1 * d1 + d2 * d2, sred);
    float inv = rsqrtf(vs * (1.0f / 384.0f) + 1e-5f);
    f16* dp = dst + (size_t)r * C;
    dp[t]       = (f16)(d0 * inv * g[t]       + b[t]);
    dp[t + 128] = (f16)(d1 * inv * g[t + 128] + b[t + 128]);
    dp[t + 256] = (f16)(d2 * inv * g[t + 256] + b[t + 256]);
}

// ---------------------------------------------------------------------------
// WMMA GEMM: OUT[M,N] = A[M,K] (f16, row-major) x Bt[N,K] (f16, row-major)
// Tile 128x64, BK=32, 256 threads = 8 waves, each wave 32x32 (2x2 frags).
// Double-buffered LDS: one barrier per K-step, global prefetch overlaps WMMA.
// EP: 0=QKV  1=PROJ(scatter+residual->f32)  2=FC1(GELU)  3=FC2(accumulate f32)
// ---------------------------------------------------------------------------
template<int EP, int N, int K>
__global__ __launch_bounds__(256)
void gemm_wmma_kernel(const f16* __restrict__ A, const f16* __restrict__ Bt,
                      const float* __restrict__ bias, f16* __restrict__ outH,
                      float* __restrict__ outF, const float* __restrict__ xres) {
    __shared__ __align__(16) f16 As[2][128 * 40];
    __shared__ __align__(16) f16 Bs[2][64 * 40];
    const int tid  = threadIdx.x;
    const int lane = tid & 31, wid = tid >> 5;
    const int wm = wid >> 1, wn = wid & 1;
    const int m0 = blockIdx.y * 128;
    const int n0 = blockIdx.x * 64;

    v8f z8 = {};
    v8f acc[2][2] = {{z8, z8}, {z8, z8}};

    const int rl = lane & 15, khalf = lane >> 4;

    // Per-thread tile-load coordinates (16B vectors)
    const int rowA0 = tid >> 2,         chA0 = tid & 3;
    const int rowA1 = (tid + 256) >> 2, chA1 = tid & 3;
    const int rowB  = tid >> 2,         chB  = tid & 3;
    const f16* gA0 = A  + (size_t)(m0 + rowA0) * K + chA0 * 8;
    const f16* gA1 = A  + (size_t)(m0 + rowA1) * K + chA1 * 8;
    const f16* gB  = Bt + (size_t)(n0 + rowB ) * K + chB  * 8;

    constexpr int STEPS = K / 32;

    // Prologue: stage tile 0
    {
        v8h ra0 = *(const v8h*)(gA0);
        v8h ra1 = *(const v8h*)(gA1);
        v8h rb  = *(const v8h*)(gB);
        *(v8h*)(As[0] + rowA0 * 40 + chA0 * 8) = ra0;
        *(v8h*)(As[0] + rowA1 * 40 + chA1 * 8) = ra1;
        *(v8h*)(Bs[0] + rowB  * 40 + chB  * 8) = rb;
    }
    __syncthreads();

    #pragma unroll 4
    for (int kt = 0; kt < STEPS; ++kt) {
        const int cur = kt & 1, nxt = cur ^ 1;
        v8h ra0, ra1, rb;
        if (kt + 1 < STEPS) {
            const int k0 = (kt + 1) * 32;
            ra0 = *(const v8h*)(gA0 + k0);
            ra1 = *(const v8h*)(gA1 + k0);
            rb  = *(const v8h*)(gB  + k0);
        }

        v16h af[2], bf[2];
        #pragma unroll
        for (int i = 0; i < 2; ++i) {
            const f16* p = As[cur] + (wm * 32 + i * 16 + rl) * 40;
            af[i] = ld16(p + khalf * 8, p + 16 + khalf * 8);
        }
        #pragma unroll
        for (int j = 0; j < 2; ++j) {
            const f16* p = Bs[cur] + (wn * 32 + j * 16 + rl) * 40;
            bf[j] = ld16(p + khalf * 8, p + 16 + khalf * 8);
        }
        #pragma unroll
        for (int i = 0; i < 2; ++i)
            #pragma unroll
            for (int j = 0; j < 2; ++j)
                acc[i][j] = WMMA_F16(af[i], bf[j], acc[i][j]);

        if (kt + 1 < STEPS) {
            *(v8h*)(As[nxt] + rowA0 * 40 + chA0 * 8) = ra0;
            *(v8h*)(As[nxt] + rowA1 * 40 + chA1 * 8) = ra1;
            *(v8h*)(Bs[nxt] + rowB  * 40 + chB  * 8) = rb;
            __syncthreads();
        }
    }

    #pragma unroll
    for (int i = 0; i < 2; ++i)
        #pragma unroll
        for (int j = 0; j < 2; ++j) {
            int nc = n0 + wn * 32 + j * 16 + rl;
            float bv = bias[nc];
            #pragma unroll
            for (int v = 0; v < 8; ++v) {
                int m = m0 + wm * 32 + i * 16 + v + ((lane >> 4) << 3);
                float val = acc[i][j][v] + bv;
                if (EP == 0) {                       // QKV: scale q columns
                    if (nc < C) val *= QSCALE;
                    outH[(size_t)m * N + nc] = (f16)val;
                } else if (EP == 1) {                // PROJ: reverse-shift scatter + residual
                    int g = swin_map(m);
                    outF[(size_t)g * C + nc] = xres[(size_t)g * C + nc] + val;
                } else if (EP == 2) {                // FC1: exact GELU
                    val = 0.5f * val * (1.0f + erff(val * 0.70710678118654752f));
                    outH[(size_t)m * N + nc] = (f16)val;
                } else {                             // FC2: accumulate into y
                    outF[(size_t)m * N + nc] += val;
                }
            }
        }
}

// ---------------------------------------------------------------------------
// Windowed attention: one block (4 waves / 128 thr) per (window, head).
// S = q k^T (one WMMA per 16x16 tile, K=HD=32) + rel-bias + shift-mask,
// softmax per row, O = P v (two K-steps of WMMA).
// ---------------------------------------------------------------------------
__global__ __launch_bounds__(128)
void attn_kernel(const f16* __restrict__ qkv, const float* __restrict__ bias_table,
                 f16* __restrict__ out) {
    __shared__ __align__(16) f16 Ps[64 * 72];
    __shared__ __align__(16) f16 Vt[32 * 72];
    __shared__ float biasS[225];
    const int tid = threadIdx.x, lane = tid & 31, w = tid >> 5;
    const int win  = blockIdx.x / NHEAD;
    const int head = blockIdx.x % NHEAD;
    const size_t base = (size_t)win * 64 * 1152;
    const int qoff = head * HD, koff = 384 + head * HD, voff = 768 + head * HD;

    // Stage this head's rel-pos bias column into LDS
    if (tid < 113)       biasS[tid]       = bias_table[tid * NHEAD + head];
    if (tid + 128 < 225) biasS[tid + 128] = bias_table[(tid + 128) * NHEAD + head];

    // Stage v^T into LDS: Vt[d][m'] = v[m'][d]
    #pragma unroll
    for (int it = 0; it < 16; ++it) {
        int lin = it * 128 + tid;
        int d = lin & 31, mp = lin >> 5;
        Vt[d * 72 + mp] = qkv[base + (size_t)mp * 1152 + voff + d];
    }

    const int rl = lane & 15, khalf = lane >> 4;

    // q fragment (rows w*16..w*16+15, K=HD=32), straight from global
    const f16* qp = qkv + base + (size_t)(w * 16 + rl) * 1152 + qoff;
    v16h aq = ld16(qp + khalf * 8, qp + 16 + khalf * 8);

    v8f s[4];
    #pragma unroll
    for (int j = 0; j < 4; ++j) {
        const f16* kp = qkv + base + (size_t)(j * 16 + rl) * 1152 + koff;
        v16h bk = ld16(kp + khalf * 8, kp + 16 + khalf * 8);
        v8f z = {};
        s[j] = WMMA_F16(aq, bk, z);
    }
    __syncthreads();   // biasS + Vt visible to all

    // relative position bias + shifted-window mask (computed arithmetically)
    const int wl = win & 63, wh = wl >> 3, ww = wl & 7;
    #pragma unroll
    for (int j = 0; j < 4; ++j) {
        const int n = j * 16 + rl;
        const int ni = n >> 3, nj = n & 7;
        const int rn = regid(wh * 8 + ni) * 3 + regid(ww * 8 + nj);
        #pragma unroll
        for (int v = 0; v < 8; ++v) {
            int m = w * 16 + v + ((lane >> 4) << 3);
            int mi = m >> 3, mj = m & 7;
            int rel = (mi - ni + 7) * 15 + (mj - nj + 7);
            float bv = biasS[rel];
            int rm = regid(wh * 8 + mi) * 3 + regid(ww * 8 + mj);
            s[j][v] += bv + ((rm != rn) ? -100.0f : 0.0f);
        }
    }

    // softmax across each 64-wide row (4 regs x 16 lanes of the half-wave)
    #pragma unroll
    for (int v = 0; v < 8; ++v) {
        float mx = fmaxf(fmaxf(s[0][v], s[1][v]), fmaxf(s[2][v], s[3][v]));
        mx = fmaxf(mx, __shfl_xor(mx, 1));
        mx = fmaxf(mx, __shfl_xor(mx, 2));
        mx = fmaxf(mx, __shfl_xor(mx, 4));
        mx = fmaxf(mx, __shfl_xor(mx, 8));
        float e0 = __expf(s[0][v] - mx), e1 = __expf(s[1][v] - mx);
        float e2 = __expf(s[2][v] - mx), e3 = __expf(s[3][v] - mx);
        float sm = e0 + e1 + e2 + e3;
        sm += __shfl_xor(sm, 1); sm += __shfl_xor(sm, 2);
        sm += __shfl_xor(sm, 4); sm += __shfl_xor(sm, 8);
        float inv = 1.0f / sm;
        int m = w * 16 + v + ((lane >> 4) << 3);
        Ps[m * 72 + 0 * 16 + rl] = (f16)(e0 * inv);
        Ps[m * 72 + 1 * 16 + rl] = (f16)(e1 * inv);
        Ps[m * 72 + 2 * 16 + rl] = (f16)(e2 * inv);
        Ps[m * 72 + 3 * 16 + rl] = (f16)(e3 * inv);
    }
    __syncthreads();

    // O = P @ v : A frags from Ps (K=64 in two chunks), B frags from Vt
    const f16* pp = Ps + (w * 16 + rl) * 72;
    v16h a0 = ld16(pp + khalf * 8,      pp + 16 + khalf * 8);
    v16h a1 = ld16(pp + 32 + khalf * 8, pp + 48 + khalf * 8);
    #pragma unroll
    for (int nb = 0; nb < 2; ++nb) {
        const f16* vp = Vt + (nb * 16 + rl) * 72;
        v16h b0 = ld16(vp + khalf * 8,      vp + 16 + khalf * 8);
        v16h b1 = ld16(vp + 32 + khalf * 8, vp + 48 + khalf * 8);
        v8f o = {};
        o = WMMA_F16(a0, b0, o);
        o = WMMA_F16(a1, b1, o);
        #pragma unroll
        for (int v = 0; v < 8; ++v) {
            int m = w * 16 + v + ((lane >> 4) << 3);
            int d = nb * 16 + rl;
            out[((size_t)win * 64 + m) * C + head * HD + d] = (f16)o[v];
        }
    }
}

// ---------------------------------------------------------------------------
extern "C" void kernel_launch(void* const* d_in, const int* in_sizes, int n_in,
                              void* d_out, int out_size, void* d_ws, size_t ws_size,
                              hipStream_t stream) {
    const float* x          = (const float*)d_in[0];
    const float* gamma1     = (const float*)d_in[1];
    const float* beta1      = (const float*)d_in[2];
    const float* w_qkv      = (const float*)d_in[3];
    const float* b_qkv      = (const float*)d_in[4];
    const float* bias_table = (const float*)d_in[5];
    const float* w_proj     = (const float*)d_in[6];
    const float* b_proj     = (const float*)d_in[7];
    const float* gamma2     = (const float*)d_in[8];
    const float* beta2      = (const float*)d_in[9];
    const float* w_fc1      = (const float*)d_in[10];
    const float* b_fc1      = (const float*)d_in[11];
    const float* w_fc2      = (const float*)d_in[12];
    const float* b_fc2      = (const float*)d_in[13];
    float* y = (float*)d_out;   // d_out doubles as the post-attention residual y

    // Workspace layout (f16 elements). hidden overlays attn_out+qkv (dead by then).
    char* ws = (char*)d_ws;
    f16* xw       = (f16*)ws;                        // [M,C]  (later reused as LN2 out)
    f16* attn_out = (f16*)(ws + 2 * S0E);            // [M,C]
    f16* qkvb     = (f16*)(ws + 4 * S0E);            // [M,3C]
    f16* hidden   = attn_out;                        // [M,HID] overlays 8*S0E bytes
    f16* wts      = (f16*)(ws + 10 * S0E);           // transposed f16 weights
    f16* wqkvT = wts;                                // [1152,384]
    f16* wprojT = wqkvT + (size_t)1152 * 384;        // [384,384]
    f16* wfc1T  = wprojT + (size_t)384 * 384;        // [1536,384]
    f16* wfc2T  = wfc1T + (size_t)1536 * 384;        // [384,1536]

    // 1) Weight transpose/convert
    transpose_f32_f16<<<(384 * 1152 + 255) / 256, 256, 0, stream>>>(w_qkv,  wqkvT, 384, 1152);
    transpose_f32_f16<<<(384 * 384  + 255) / 256, 256, 0, stream>>>(w_proj, wprojT, 384, 384);
    transpose_f32_f16<<<(384 * 1536 + 255) / 256, 256, 0, stream>>>(w_fc1,  wfc1T, 384, 1536);
    transpose_f32_f16<<<(1536 * 384 + 255) / 256, 256, 0, stream>>>(w_fc2,  wfc2T, 1536, 384);

    // 2) LN1 + shift + window partition -> f16
    ln_kernel<<<MROWS, 128, 0, stream>>>(x, gamma1, beta1, xw, 1);

    // 3) QKV GEMM (q scaled by HD^-1/2)
    gemm_wmma_kernel<0, 1152, 384><<<dim3(1152 / 64, MROWS / 128), 256, 0, stream>>>(
        xw, wqkvT, b_qkv, qkvb, nullptr, nullptr);

    // 4) Windowed attention (2048 windows x 12 heads)
    attn_kernel<<<2048 * NHEAD, 128, 0, stream>>>(qkvb, bias_table, attn_out);

    // 5) Proj GEMM + window-reverse + un-shift + residual -> y (f32 in d_out)
    gemm_wmma_kernel<1, 384, 384><<<dim3(384 / 64, MROWS / 128), 256, 0, stream>>>(
        attn_out, wprojT, b_proj, nullptr, y, x);

    // 6) LN2 -> f16 (reuse xw region)
    ln_kernel<<<MROWS, 128, 0, stream>>>(y, gamma2, beta2, xw, 0);

    // 7) FC1 + GELU
    gemm_wmma_kernel<2, 1536, 384><<<dim3(1536 / 64, MROWS / 128), 256, 0, stream>>>(
        xw, wfc1T, b_fc1, hidden, nullptr, nullptr);

    // 8) FC2, accumulate into y -> final output
    gemm_wmma_kernel<3, 384, 1536><<<dim3(384 / 64, MROWS / 128), 256, 0, stream>>>(
        hidden, wfc2T, b_fc2, nullptr, y, nullptr);
}
